// GraphSNN_41686952575157
// MI455X (gfx1250) — matrix operand order repaired and checked by reading, not compiled, not run on hardware
//
#include <hip/hip_runtime.h>

// ---------------------------------------------------------------------------
// CDNA5 (gfx1250) fused GraphSNN forward: bf16 WMMA 16x16x32, f32 accumulate.
// Each wave computes a 64x16 column (4 M-tiles) per nt so one B fragment
// feeds 4 WMMAs (4x less L2 weight traffic, 4 independent XDL chains).
// ---------------------------------------------------------------------------

typedef __attribute__((ext_vector_type(16))) __bf16       v16bf;
typedef __attribute__((ext_vector_type(8)))  float        v8f;
typedef __attribute__((ext_vector_type(4)))  unsigned int v4u;

union FragB {
  v16bf bf;
  v4u   u4[2];
};

#define LSTRIDE   520   // 512 bf16 + 8 pad -> row stride 1040B (bank-conflict-free b128)
#define TILE_ROWS 64
#define NWAVES    8

__device__ __forceinline__ unsigned short f2bf(float x) {
  unsigned int u = __float_as_uint(x);
  u += 0x7FFFu + ((u >> 16) & 1u);   // round-to-nearest-even
  return (unsigned short)(u >> 16);
}

__device__ __forceinline__ v8f splat8(float v) {
  v8f c = {v, v, v, v, v, v, v, v};
  return c;
}

// leaky_relu(x) = max(x, 0.01*x) since 0.01 < 1  (no cmp/cndmask)
__device__ __forceinline__ void lrelu8(v8f& c) {
#pragma unroll
  for (int r = 0; r < 8; ++r) c[r] = fmaxf(c[r], 0.01f * c[r]);
}

// 64x16 output column: c[m] = A[mt=m] @ Wp[:,nt] + bias[nt], m = 0..3.
// A from LDS (bf16, stride LSTRIDE), B from fragment-packed global weights.
// Per kt: 1 B fragment (2x global_load_b128) shared by 4 WMMAs.
__device__ __forceinline__ void quad_mm(const unsigned short* sIn,
                                        const unsigned short* __restrict__ Wp,
                                        const float* __restrict__ bias,
                                        int K, int nt, int lane, v8f c[4]) {
  const int Ktiles = K >> 5;
  float bv = bias[(nt << 4) + (lane & 15)];
#pragma unroll
  for (int m = 0; m < 4; ++m) c[m] = splat8(bv);
  // A fragment addressing per ISA 16-bit A 16x32 layout:
  // lanes 0-15: row=m0+lane, K chunks [k0..k0+7],[k0+16..k0+23]
  // lanes 16-31: same rows, chunks shifted by +8
  const unsigned short* aBase =
      sIn + (lane & 15) * LSTRIDE + ((lane >> 4) << 3);
  const unsigned short* bPtr =
      Wp + (((unsigned)(nt * Ktiles)) << 9) + (lane << 4);
  for (int kt = 0; kt < Ktiles; ++kt) {
    FragB b;
    b.u4[0] = *(const v4u*)(bPtr);
    b.u4[1] = *(const v4u*)(bPtr + 8);
    bPtr += 512;
    const unsigned short* ak = aBase + (kt << 5);
#pragma unroll
    for (int m = 0; m < 4; ++m) {
      FragB a;
      const unsigned short* am = ak + m * (16 * LSTRIDE);
      a.u4[0] = *(const v4u*)(am);
      a.u4[1] = *(const v4u*)(am + 16);
      c[m] = __builtin_amdgcn_wmma_f32_16x16x32_bf16(false, a.bf, false, b.bf,
                                                     (short)0, c[m], false, false);
    }
  }
}

// Store a 16x16 f32 accumulator tile into LDS as bf16 (C layout: VGPR r,
// lane l -> M = r + 8*(l>>4), N = l&15).
__device__ __forceinline__ void store_tile_bf16(unsigned short* sOut, const v8f& c,
                                                int mt, int nt, int lane) {
  int col  = (nt << 4) + (lane & 15);
  int row0 = (mt << 4) + ((lane >> 4) << 3);
#pragma unroll
  for (int r = 0; r < 8; ++r)
    sOut[(row0 + r) * LSTRIDE + col] = f2bf(c[r]);
}

// ---------------------------------------------------------------------------
// Weight packing: f32 row-major [K,N] -> bf16 WMMA-B fragment tiles.
// dst[(nt*Ktiles + kt)*512 + lane*16 + i] = W[kt*32 + (i>>3)*16 + (lane>>4)*8 + (i&7)][nt*16 + (lane&15)]
// ---------------------------------------------------------------------------
__global__ void pack_weights(const float* __restrict__ W, unsigned short* __restrict__ dst,
                             int K, int N) {
  int total = K * N;
  int Ktiles = K >> 5;
  for (int e = blockIdx.x * blockDim.x + threadIdx.x; e < total;
       e += gridDim.x * blockDim.x) {
    int tile = e >> 9;
    int rem  = e & 511;
    int lane = rem >> 4;
    int i    = rem & 15;
    int nt = tile / Ktiles;
    int kt = tile - nt * Ktiles;
    int k = (kt << 5) + ((i >> 3) << 4) + ((lane >> 4) << 3) + (i & 7);
    int n = (nt << 4) + (lane & 15);
    dst[e] = f2bf(W[k * N + n]);
  }
}

__global__ void zero_kernel(float* __restrict__ p, int n) {
  for (int i = blockIdx.x * blockDim.x + threadIdx.x; i < n;
       i += gridDim.x * blockDim.x)
    p[i] = 0.f;
}

// ---------------------------------------------------------------------------
// Node stack: s = lrelu(lrelu(lrelu(x@W1+b1)@W2+b2)@W3+b3); segment_sum -> s1
// 64-row tiles, activations LDS-resident, atomics for segment sum.
// ---------------------------------------------------------------------------
__global__ __launch_bounds__(256) void node_mlp_kernel(
    const float* __restrict__ X, const int* __restrict__ dag_ids,
    const unsigned short* __restrict__ W1p, const unsigned short* __restrict__ W2p,
    const unsigned short* __restrict__ W3p,
    const float* __restrict__ b1, const float* __restrict__ b2,
    const float* __restrict__ b3,
    float* __restrict__ s1, int Nrows) {
  __shared__ unsigned short bufA[TILE_ROWS * LSTRIDE];
  __shared__ unsigned short bufB[TILE_ROWS * LSTRIDE];
  __shared__ int sdag[TILE_ROWS];

  int tid  = threadIdx.x;
  int wave = tid >> 5;
  int lane = tid & 31;
  int rowBase = blockIdx.x * TILE_ROWS;

  if (tid < TILE_ROWS) {
    int gr = rowBase + tid;
    sdag[tid] = (gr < Nrows) ? dag_ids[gr] : -1;
  }
  // Load x tile [64 x 128] f32 -> bf16 LDS (coalesced)
  for (int e = tid; e < TILE_ROWS * 128; e += 256) {
    int r = e >> 7, cc = e & 127;
    int gr = rowBase + r;
    float v = (gr < Nrows) ? X[gr * 128 + cc] : 0.f;
    bufA[r * LSTRIDE + cc] = f2bf(v);
  }
  __syncthreads();

  // Layer 1: K=128, N=512, bufA -> bufB
  for (int nt = wave; nt < 32; nt += NWAVES) {
    v8f c[4];
    quad_mm(bufA, W1p, b1, 128, nt, lane, c);
#pragma unroll
    for (int m = 0; m < 4; ++m) {
      lrelu8(c[m]);
      store_tile_bf16(bufB, c[m], m, nt, lane);
    }
  }
  __syncthreads();

  // Layer 2: K=512, N=512, bufB -> bufA
  for (int nt = wave; nt < 32; nt += NWAVES) {
    v8f c[4];
    quad_mm(bufB, W2p, b2, 512, nt, lane, c);
#pragma unroll
    for (int m = 0; m < 4; ++m) {
      lrelu8(c[m]);
      store_tile_bf16(bufA, c[m], m, nt, lane);
    }
  }
  __syncthreads();

  // Layer 3: K=512, N=256, bufA -> segment-sum atomics into s1
  for (int nt = wave; nt < 16; nt += NWAVES) {
    v8f c[4];
    quad_mm(bufA, W3p, b3, 512, nt, lane, c);
    int col = (nt << 4) + (lane & 15);
#pragma unroll
    for (int m = 0; m < 4; ++m) {
      lrelu8(c[m]);
      int lrow0 = (m << 4) + ((lane >> 4) << 3);
#pragma unroll
      for (int r = 0; r < 8; ++r) {
        int d = sdag[lrow0 + r];
        if (d >= 0) atomicAdd(&s1[d * 256 + col], c[m][r]);
      }
    }
  }
}

// ---------------------------------------------------------------------------
// Global stack: g = lrelu(lrelu(lrelu(s1@W4+b4)@W5+b5)@W6+b6); s2 = mask @ g
// ---------------------------------------------------------------------------
__global__ __launch_bounds__(256) void global_mlp_kernel(
    const float* __restrict__ S1, const float* __restrict__ mask,
    const unsigned short* __restrict__ W4p, const unsigned short* __restrict__ W5p,
    const unsigned short* __restrict__ W6p,
    const float* __restrict__ b4, const float* __restrict__ b5,
    const float* __restrict__ b6,
    float* __restrict__ s2out, int Ndags) {
  __shared__ unsigned short bufA[TILE_ROWS * LSTRIDE];
  __shared__ unsigned short bufB[TILE_ROWS * LSTRIDE];
  __shared__ float smask[TILE_ROWS];
  __shared__ float acc[256];

  int tid  = threadIdx.x;
  int wave = tid >> 5;
  int lane = tid & 31;
  int rowBase = blockIdx.x * TILE_ROWS;

  if (tid < TILE_ROWS) {
    int gr = rowBase + tid;
    smask[tid] = (gr < Ndags) ? mask[gr] : 0.f;
  }
  acc[tid] = 0.f;
  // Load s1 tile [64 x 256] f32 -> bf16 LDS
  for (int e = tid; e < TILE_ROWS * 256; e += 256) {
    int r = e >> 8, cc = e & 255;
    int gr = rowBase + r;
    float v = (gr < Ndags) ? S1[gr * 256 + cc] : 0.f;
    bufA[r * LSTRIDE + cc] = f2bf(v);
  }
  __syncthreads();

  // Layer 4: K=256, N=512, bufA -> bufB
  for (int nt = wave; nt < 32; nt += NWAVES) {
    v8f c[4];
    quad_mm(bufA, W4p, b4, 256, nt, lane, c);
#pragma unroll
    for (int m = 0; m < 4; ++m) {
      lrelu8(c[m]);
      store_tile_bf16(bufB, c[m], m, nt, lane);
    }
  }
  __syncthreads();

  // Layer 5: K=512, N=512, bufB -> bufA
  for (int nt = wave; nt < 32; nt += NWAVES) {
    v8f c[4];
    quad_mm(bufB, W5p, b5, 512, nt, lane, c);
#pragma unroll
    for (int m = 0; m < 4; ++m) {
      lrelu8(c[m]);
      store_tile_bf16(bufA, c[m], m, nt, lane);
    }
  }
  __syncthreads();

  // Layer 6: K=512, N=256, bufA -> mask-weighted LDS reduction
  for (int nt = wave; nt < 16; nt += NWAVES) {
    v8f c[4];
    quad_mm(bufA, W6p, b6, 512, nt, lane, c);
    int col = (nt << 4) + (lane & 15);
#pragma unroll
    for (int m = 0; m < 4; ++m) {
      lrelu8(c[m]);
      int lrow0 = (m << 4) + ((lane >> 4) << 3);
#pragma unroll
      for (int r = 0; r < 8; ++r)
        atomicAdd(&acc[col], smask[lrow0 + r] * c[m][r]);
    }
  }
  __syncthreads();
  atomicAdd(&s2out[tid], acc[tid]);
}

// ---------------------------------------------------------------------------
// Launch
// ---------------------------------------------------------------------------
extern "C" void kernel_launch(void* const* d_in, const int* in_sizes, int n_in,
                              void* d_out, int out_size, void* d_ws, size_t ws_size,
                              hipStream_t stream) {
  const float* X    = (const float*)d_in[0];
  const int*   dag  = (const int*)d_in[1];
  const float* mask = (const float*)d_in[2];
  const float* W1 = (const float*)d_in[3];  const float* b1 = (const float*)d_in[4];
  const float* W2 = (const float*)d_in[5];  const float* b2 = (const float*)d_in[6];
  const float* W3 = (const float*)d_in[7];  const float* b3 = (const float*)d_in[8];
  const float* W4 = (const float*)d_in[9];  const float* b4 = (const float*)d_in[10];
  const float* W5 = (const float*)d_in[11]; const float* b5 = (const float*)d_in[12];
  const float* W6 = (const float*)d_in[13]; const float* b6 = (const float*)d_in[14];

  const int N     = in_sizes[0] / 128;  // 100000
  const int NDAGS = in_sizes[2];        // 2000
  float* out = (float*)d_out;
  float* s1  = out;                     // [NDAGS, 256]
  float* s2  = out + NDAGS * 256;       // [256]

  unsigned short* ws  = (unsigned short*)d_ws;
  unsigned short* W1p = ws;             // 128*512
  unsigned short* W2p = W1p + 128 * 512;
  unsigned short* W3p = W2p + 512 * 512;
  unsigned short* W4p = W3p + 512 * 256;
  unsigned short* W5p = W4p + 256 * 512;
  unsigned short* W6p = W5p + 512 * 512;

  zero_kernel<<<512, 256, 0, stream>>>(out, NDAGS * 256 + 256);
  pack_weights<<<256, 256, 0, stream>>>(W1, W1p, 128, 512);
  pack_weights<<<256, 256, 0, stream>>>(W2, W2p, 512, 512);
  pack_weights<<<256, 256, 0, stream>>>(W3, W3p, 512, 256);
  pack_weights<<<256, 256, 0, stream>>>(W4, W4p, 256, 512);
  pack_weights<<<256, 256, 0, stream>>>(W5, W5p, 512, 512);
  pack_weights<<<256, 256, 0, stream>>>(W6, W6p, 512, 256);

  int nodeBlocks = (N + TILE_ROWS - 1) / TILE_ROWS;
  node_mlp_kernel<<<nodeBlocks, 256, 0, stream>>>(X, dag, W1p, W2p, W3p,
                                                  b1, b2, b3, s1, N);
  int dagBlocks = (NDAGS + TILE_ROWS - 1) / TILE_ROWS;
  global_mlp_kernel<<<dagBlocks, 256, 0, stream>>>(s1, mask, W4p, W5p, W6p,
                                                   b4, b5, b6, s2, NDAGS);
}
